// CodeEncoderLayer_14851996909621
// MI455X (gfx1250) — matrix operand order repaired and checked by reading, not compiled, not run on hardware
//
#include <hip/hip_runtime.h>

// ---------------------------------------------------------------------------
// CodeEncoderLayer on MI455X (gfx1250): fused transformer encoder layer.
// - All GEMMs: V_WMMA_F32_16X16X32_BF16, 32x64 wave tiles, ping-pong
//   register pipeline (unroll-by-2, no buffer copies), launch_bounds(256,1)
//   so the allocator can hold 8 accumulators + 2 fragment sets without spills.
// - Attention: flash-style, K tiles staged to LDS via GLOBAL_LOAD_ASYNC_TO_LDS
//   (double-buffered, s_wait_asynccnt), bias/masks applied on the fly.
// ---------------------------------------------------------------------------

#define DEVINL __device__ __forceinline__

typedef __attribute__((ext_vector_type(16))) __bf16 v16bf;
typedef __attribute__((ext_vector_type(8)))  float  v8f;

namespace cel {
constexpr int E  = 512;     // embed
constexpr int H  = 8;       // heads
constexpr int F  = 2048;    // ffn
constexpr int Ns = 1024;    // sequence (N)
constexpr int Bb = 4;       // batch
constexpr int HD = 64;      // head dim
constexpr int M  = Ns * Bb; // 4096 flattened rows (row m = n*B + b)
}

union FragU { v16bf v; uint4 u[2]; };

DEVINL v8f vzero() {
  v8f z;
#pragma unroll
  for (int i = 0; i < 8; ++i) z[i] = 0.0f;
  return z;
}

// A fragment (16x32, M x K): lane holds row (lane&15), K = {kb..kb+7, kb+16..kb+23}
DEVINL v16bf load_a_frag(const __bf16* base, int ld, int lane) {
  const int row = lane & 15;
  const int kb  = (lane >> 4) << 3;
  const __bf16* p = base + row * ld + kb;
  FragU f;
  f.u[0] = *(const uint4*)(p);
  f.u[1] = *(const uint4*)(p + 16);
  return f.v;
}

// B fragment (32x16, K x N): lane holds column (lane&15), K = {kb..kb+7, kb+16..kb+23}, kb = (lane>>4)*8.
// base points at row-major [Ncols x K] storage of B^T (e.g. weight rows), ld = K stride.
DEVINL v16bf load_b_frag(const __bf16* base, int ld, int lane) {
  const int row = lane & 15;
  const int kb  = (lane >> 4) << 3;
  const __bf16* p = base + row * ld + kb;
  FragU f;
  f.u[0] = *(const uint4*)(p);
  f.u[1] = *(const uint4*)(p + 16);
  return f.v;
}

DEVINL v8f wmma_bf16(v16bf a, v16bf b, v8f c) {
  return __builtin_amdgcn_wmma_f32_16x16x32_bf16(false, a, false, b, (short)0, c,
                                                 false, false);
}

// ---- CDNA5 async / wait helpers -------------------------------------------
DEVINL void async_copy_b128(unsigned lds_addr, unsigned long long gaddr) {
  asm volatile("global_load_async_to_lds_b128 %0, %1, off"
               :: "v"(lds_addr), "v"(gaddr) : "memory");
}
DEVINL void wait_async_le(void) { asm volatile("s_wait_asynccnt 0x8" ::: "memory"); }
DEVINL void wait_async_0(void)  { asm volatile("s_wait_asynccnt 0x0" ::: "memory"); }
DEVINL void wait_ds_0(void)     { asm volatile("s_wait_dscnt 0x0"    ::: "memory"); }

// stage 32 keys x 64 bf16 (4 KB, contiguous) into LDS: 8 x b128 per lane-set
DEVINL void stage_tile4k(__bf16* dst, const __bf16* src, int lane) {
  unsigned lbase = (unsigned)(unsigned long long)(uintptr_t)dst + lane * 16;
  unsigned long long gbase = (unsigned long long)(uintptr_t)src + lane * 16;
#pragma unroll
  for (int i = 0; i < 8; ++i)
    async_copy_b128(lbase + i * 512, gbase + i * 512);
}

// ---------------------------------------------------------------------------
// fp32 -> bf16 conversion (grid-stride)
// ---------------------------------------------------------------------------
__global__ void cvt_bf16_kernel(const float* __restrict__ in, __bf16* __restrict__ out, int n) {
  int i = blockIdx.x * blockDim.x + threadIdx.x;
  int stride = gridDim.x * blockDim.x;
  for (; i < n; i += stride) out[i] = (__bf16)in[i];
}

// ---------------------------------------------------------------------------
// pbias/cbias: (B,H,N) = x @ Wp^T + bp  (tiny GEMV; VALU is fine)
// ---------------------------------------------------------------------------
__global__ void pcbias_kernel(const float* __restrict__ x,
                              const float* __restrict__ Wp, const float* __restrict__ bp,
                              const float* __restrict__ Wc, const float* __restrict__ bc,
                              float* __restrict__ pbias, float* __restrict__ cbias) {
  using namespace cel;
  int idx = blockIdx.x * blockDim.x + threadIdx.x;   // over B*H*Ns
  int n  = idx % Ns;
  int bh = idx / Ns;
  int h  = bh & (H - 1);
  int b  = bh >> 3;
  const float* xr  = x  + (n * Bb + b) * E;
  const float* wpr = Wp + h * E;
  const float* wcr = Wc + h * E;
  float sp = 0.f, sc = 0.f;
  for (int e = 0; e < E; e += 4) {
    float4 xv = *(const float4*)(xr + e);
    float4 wp = *(const float4*)(wpr + e);
    float4 wc = *(const float4*)(wcr + e);
    sp += xv.x * wp.x + xv.y * wp.y + xv.z * wp.z + xv.w * wp.w;
    sc += xv.x * wc.x + xv.y * wc.y + xv.z * wc.z + xv.w * wc.w;
  }
  pbias[idx] = sp + bp[h];
  cbias[idx] = sc + bc[h];
}

// ---------------------------------------------------------------------------
// WMMA tile loop: one wave computes a 32 x 64 tile (2x4 accumulators).
// Ping-pong pipeline unrolled by 2 (set0/set1), no register copies:
//   [load set1 @k+32] [wmma set0 @k] [load set0 @k+64] [wmma set1 @k+32]
// K must be a multiple of 64 (512 / 2048 here).
// ---------------------------------------------------------------------------
DEVINL void gemm_tile32x64(const __bf16* __restrict__ A, const __bf16* __restrict__ Bw,
                           int K, int m0, int n0, int lane, v8f acc[2][4]) {
  const __bf16* a0p = A + (size_t)m0 * K;
  const __bf16* a1p = A + (size_t)(m0 + 16) * K;
  const __bf16* bp  = Bw + (size_t)n0 * K;

  v16bf a0 = load_a_frag(a0p, K, lane);
  v16bf a1 = load_a_frag(a1p, K, lane);
  v16bf b0[4];
#pragma unroll
  for (int t = 0; t < 4; ++t) b0[t] = load_b_frag(bp + (size_t)t * 16 * K, K, lane);

  int k = 0;
  for (; k + 96 <= K; k += 64) {
    // set1 <- k+32
    v16bf c0 = load_a_frag(a0p + k + 32, K, lane);
    v16bf c1 = load_a_frag(a1p + k + 32, K, lane);
    v16bf d0[4];
#pragma unroll
    for (int t = 0; t < 4; ++t)
      d0[t] = load_b_frag(bp + (size_t)t * 16 * K + k + 32, K, lane);
    // consume set0 (k)
#pragma unroll
    for (int t = 0; t < 4; ++t) {
      acc[0][t] = wmma_bf16(a0, b0[t], acc[0][t]);
      acc[1][t] = wmma_bf16(a1, b0[t], acc[1][t]);
    }
    // set0 <- k+64
    a0 = load_a_frag(a0p + k + 64, K, lane);
    a1 = load_a_frag(a1p + k + 64, K, lane);
#pragma unroll
    for (int t = 0; t < 4; ++t)
      b0[t] = load_b_frag(bp + (size_t)t * 16 * K + k + 64, K, lane);
    // consume set1 (k+32)
#pragma unroll
    for (int t = 0; t < 4; ++t) {
      acc[0][t] = wmma_bf16(c0, d0[t], acc[0][t]);
      acc[1][t] = wmma_bf16(c1, d0[t], acc[1][t]);
    }
  }
  // tail: 64 K-values left; set0 holds k, load set1 @k+32, consume both
  {
    v16bf c0 = load_a_frag(a0p + k + 32, K, lane);
    v16bf c1 = load_a_frag(a1p + k + 32, K, lane);
    v16bf d0[4];
#pragma unroll
    for (int t = 0; t < 4; ++t)
      d0[t] = load_b_frag(bp + (size_t)t * 16 * K + k + 32, K, lane);
#pragma unroll
    for (int t = 0; t < 4; ++t) {
      acc[0][t] = wmma_bf16(a0, b0[t], acc[0][t]);
      acc[1][t] = wmma_bf16(a1, b0[t], acc[1][t]);
    }
#pragma unroll
    for (int t = 0; t < 4; ++t) {
      acc[0][t] = wmma_bf16(c0, d0[t], acc[0][t]);
      acc[1][t] = wmma_bf16(c1, d0[t], acc[1][t]);
    }
  }
}

// ---------------------------------------------------------------------------
// QKV projection: (M x E) x (E x 3E) -> scatter q,k as (B,H,N,HD) and v
// transposed as (B,H,HD,N).  grid: (M/32, 3), block 256 (8 waves)
// ---------------------------------------------------------------------------
__global__ void __launch_bounds__(256, 1)
gemm_qkv_kernel(const __bf16* __restrict__ A, const __bf16* __restrict__ Bw,
                const float* __restrict__ bias,
                __bf16* __restrict__ qout, __bf16* __restrict__ kout,
                __bf16* __restrict__ vtout) {
  using namespace cel;
  const int lane = threadIdx.x & 31;
  const int wave = threadIdx.x >> 5;
  const int m0 = blockIdx.x * 32;
  const int n0 = (blockIdx.y * 8 + wave) * 64;
  v8f acc[2][4] = {{vzero(), vzero(), vzero(), vzero()},
                   {vzero(), vzero(), vzero(), vzero()}};
  gemm_tile32x64(A, Bw, E, m0, n0, lane, acc);

  const int mh = 8 * (lane >> 4);
  const int nl = lane & 15;
#pragma unroll
  for (int mi = 0; mi < 2; ++mi) {
#pragma unroll
    for (int t = 0; t < 4; ++t) {
#pragma unroll
      for (int v = 0; v < 8; ++v) {
        int m = m0 + mi * 16 + mh + v;  // row = n*B + b
        int j = n0 + t * 16 + nl;       // column in [0, 3E)
        float val = acc[mi][t][v] + bias[j];
        int nq = m >> 2, b = m & 3;     // B = 4
        int sel = j >> 9;               // 0=q 1=k 2=v
        int h = (j >> 6) & 7, d = j & 63;
        int bh = b * H + h;
        __bf16 bv = (__bf16)val;
        if (sel == 0)      qout[(bh * Ns + nq) * HD + d] = bv;
        else if (sel == 1) kout[(bh * Ns + nq) * HD + d] = bv;
        else               vtout[(bh * HD + d) * Ns + nq] = bv;
      }
    }
  }
}

// ---------------------------------------------------------------------------
// f32-output GEMM with bias: Cout[m*Nout+j] = A@Bw^T + bias   (Wo, FFN2)
// grid: (M/32, Nout/512), block 256
// ---------------------------------------------------------------------------
__global__ void __launch_bounds__(256, 1)
gemm_f32_kernel(const __bf16* __restrict__ A, const __bf16* __restrict__ Bw,
                const float* __restrict__ bias, float* __restrict__ Cout,
                int K, int Nout) {
  const int lane = threadIdx.x & 31;
  const int wave = threadIdx.x >> 5;
  const int m0 = blockIdx.x * 32;
  const int n0 = (blockIdx.y * 8 + wave) * 64;
  v8f acc[2][4] = {{vzero(), vzero(), vzero(), vzero()},
                   {vzero(), vzero(), vzero(), vzero()}};
  gemm_tile32x64(A, Bw, K, m0, n0, lane, acc);

  const int mh = 8 * (lane >> 4);
  const int nl = lane & 15;
#pragma unroll
  for (int mi = 0; mi < 2; ++mi) {
#pragma unroll
    for (int t = 0; t < 4; ++t) {
#pragma unroll
      for (int v = 0; v < 8; ++v) {
        int j = n0 + t * 16 + nl;
        Cout[(size_t)(m0 + mi * 16 + mh + v) * Nout + j] = acc[mi][t][v] + bias[j];
      }
    }
  }
}

// ---------------------------------------------------------------------------
// FFN1: relu(y @ W1^T + b1) -> bf16.  grid: (M/32, F/512), block 256
// ---------------------------------------------------------------------------
__global__ void __launch_bounds__(256, 1)
gemm_relu_kernel(const __bf16* __restrict__ A, const __bf16* __restrict__ Bw,
                 const float* __restrict__ bias, __bf16* __restrict__ Cout) {
  using namespace cel;
  const int lane = threadIdx.x & 31;
  const int wave = threadIdx.x >> 5;
  const int m0 = blockIdx.x * 32;
  const int n0 = (blockIdx.y * 8 + wave) * 64;
  v8f acc[2][4] = {{vzero(), vzero(), vzero(), vzero()},
                   {vzero(), vzero(), vzero(), vzero()}};
  gemm_tile32x64(A, Bw, E, m0, n0, lane, acc);

  const int mh = 8 * (lane >> 4);
  const int nl = lane & 15;
#pragma unroll
  for (int mi = 0; mi < 2; ++mi) {
#pragma unroll
    for (int t = 0; t < 4; ++t) {
#pragma unroll
      for (int v = 0; v < 8; ++v) {
        int j = n0 + t * 16 + nl;
        float val = acc[mi][t][v] + bias[j];
        Cout[(size_t)(m0 + mi * 16 + mh + v) * F + j] = (__bf16)fmaxf(val, 0.0f);
      }
    }
  }
}

// ---------------------------------------------------------------------------
// Fused flash attention with on-the-fly additive bias + masks.
// K tiles are staged into LDS with GLOBAL_LOAD_ASYNC_TO_LDS (double-buffered).
// grid: (Ns/(16*8), B*H), block 256 (8 waves; each wave owns a 16-query tile)
// ---------------------------------------------------------------------------
__global__ void attn_kernel(const __bf16* __restrict__ qb, const __bf16* __restrict__ kb,
                            const __bf16* __restrict__ vt,
                            const float* __restrict__ pbias, const float* __restrict__ cbias,
                            const float* __restrict__ pm,
                            const unsigned char* __restrict__ hidden,
                            const unsigned char* __restrict__ pad,
                            __bf16* __restrict__ ctxb) {
  using namespace cel;
  __shared__ __bf16 Kbuf[8][2][32][64];  // per-wave double-buffered K tiles (64 KB)
  __shared__ __bf16 Pbuf[8][16][32];     // per-wave probability tile (8 KB)

  const int lane = threadIdx.x & 31;
  const int wave = threadIdx.x >> 5;
  const int bh = blockIdx.y;
  const int b = bh >> 3, h = bh & 7;
  const int q0 = (blockIdx.x * 8 + wave) * 16;

  const __bf16* qbase = qb + (bh * Ns + q0) * HD;
  const __bf16* kbase = kb + bh * Ns * HD;
  const __bf16* vbase = vt + bh * HD * Ns;
  const float* pmB = pm + (size_t)b * Ns * Ns;
  const unsigned char* hidB = hidden + (size_t)b * Ns * Ns;
  const unsigned char* padB = pad + b * Ns;

  // q fragments (K = HD = 64 -> two 16x32 chunks), loop invariant
  v16bf aq0 = load_a_frag(qbase, HD, lane);
  v16bf aq1 = load_a_frag(qbase + 32, HD, lane);

  const int mh = 8 * (lane >> 4);
  const int nl = lane & 15;
  float pb[8], cb[8], rmax[8], rsum[8];
#pragma unroll
  for (int v = 0; v < 8; ++v) {
    pb[v]   = pbias[bh * Ns + q0 + mh + v];
    cb[v]   = cbias[bh * Ns + q0 + mh + v];
    rmax[v] = -1e30f;
    rsum[v] = 0.0f;
  }
  v8f acc[4] = {vzero(), vzero(), vzero(), vzero()};

  // prefetch first K chunk into buffer 0
  stage_tile4k(&Kbuf[wave][0][0][0], kbase, lane);

  for (int k0 = 0; k0 < Ns; k0 += 32) {
    const int cur = (k0 >> 5) & 1;
    // all LDS reads of the buffer being overwritten finished last iteration
    wait_ds_0();
    if (k0 + 32 < Ns) {
      stage_tile4k(&Kbuf[wave][cur ^ 1][0][0], kbase + (k0 + 32) * HD, lane);
      wait_async_le();   // <=8 outstanding: current buffer's 8 copies landed
    } else {
      wait_async_0();
    }
    const __bf16* kl = &Kbuf[wave][cur][0][0];

    // ---- scores: two 16x16 tiles (32 keys), K from LDS --------------------
    v8f s[2] = {vzero(), vzero()};
#pragma unroll
    for (int t = 0; t < 2; ++t) {
      v16bf bk0 = load_b_frag(kl + (t * 16) * HD, HD, lane);
      v16bf bk1 = load_b_frag(kl + (t * 16) * HD + 32, HD, lane);
      s[t] = wmma_bf16(aq0, bk0, s[t]);
      s[t] = wmma_bf16(aq1, bk1, s[t]);
    }
    // ---- scale + additive bias + masks ------------------------------------
    float p[2][8];
#pragma unroll
    for (int t = 0; t < 2; ++t) {
      int kg = k0 + t * 16 + nl;
      bool padk = padB[kg] != 0;
#pragma unroll
      for (int v = 0; v < 8; ++v) {
        int qg = q0 + mh + v;
        float sc = s[t][v] * 0.125f
                 + pmB[(size_t)kg * Ns + qg] * pb[v]
                 + pmB[(size_t)qg * Ns + kg] * cb[v];
        if (padk || hidB[(size_t)qg * Ns + kg]) sc = -1e30f;
        p[t][v] = sc;
      }
    }
    // ---- online softmax (row reductions across 16-lane groups) ------------
#pragma unroll
    for (int v = 0; v < 8; ++v) {
      float cm = fmaxf(p[0][v], p[1][v]);
      cm = fmaxf(cm, __shfl_xor(cm, 1, 32));
      cm = fmaxf(cm, __shfl_xor(cm, 2, 32));
      cm = fmaxf(cm, __shfl_xor(cm, 4, 32));
      cm = fmaxf(cm, __shfl_xor(cm, 8, 32));
      float mnew = fmaxf(rmax[v], cm);
      float corr = __expf(rmax[v] - mnew);
      rmax[v] = mnew;
      float p0 = __expf(p[0][v] - mnew);
      float p1 = __expf(p[1][v] - mnew);
      p[0][v] = p0;
      p[1][v] = p1;
      float rs = p0 + p1;
      rs += __shfl_xor(rs, 1, 32);
      rs += __shfl_xor(rs, 2, 32);
      rs += __shfl_xor(rs, 4, 32);
      rs += __shfl_xor(rs, 8, 32);
      rsum[v] = rsum[v] * corr + rs;
#pragma unroll
      for (int t2 = 0; t2 < 4; ++t2) acc[t2][v] *= corr;
    }
    // ---- P through LDS (C-layout -> A-fragment layout) --------------------
#pragma unroll
    for (int t = 0; t < 2; ++t)
#pragma unroll
      for (int v = 0; v < 8; ++v)
        Pbuf[wave][mh + v][t * 16 + nl] = (__bf16)p[t][v];
    wait_ds_0();   // same-wave LDS RAW
    v16bf ap = load_a_frag(&Pbuf[wave][0][0], 32, lane);
    // ---- acc += P @ V (V^T rows are contiguous in keys) -------------------
#pragma unroll
    for (int t2 = 0; t2 < 4; ++t2) {
      v16bf bv = load_b_frag(vbase + (t2 * 16) * Ns + k0, Ns, lane);
      acc[t2] = wmma_bf16(ap, bv, acc[t2]);
    }
  }
  // ---- normalize and scatter context into (N,B,E) row-major bf16 ----------
#pragma unroll
  for (int v = 0; v < 8; ++v) {
    float inv = 1.0f / rsum[v];
    int nq = q0 + mh + v;
#pragma unroll
    for (int t2 = 0; t2 < 4; ++t2) {
      int d = t2 * 16 + nl;
      ctxb[(nq * Bb + b) * E + h * HD + d] = (__bf16)(acc[t2][v] * inv);
    }
  }
}

// ---------------------------------------------------------------------------
// LayerNorm(residual_a + residual_b): writes f32 (and optional bf16 copy)
// grid: M rows, block 256 (each thread owns 2 of the 512 elements)
// ---------------------------------------------------------------------------
__global__ void ln_kernel(const float* __restrict__ xa, const float* __restrict__ xb,
                          const float* __restrict__ g, const float* __restrict__ be,
                          float* __restrict__ yf, __bf16* __restrict__ ybf) {
  using namespace cel;
  __shared__ float sbuf[8], qbuf[8];
  const int row = blockIdx.x;
  const int tid = threadIdx.x;
  const size_t base = (size_t)row * E;
  float t0 = xa[base + tid] + xb[base + tid];
  float t1 = xa[base + tid + 256] + xb[base + tid + 256];
  float s = t0 + t1, q = t0 * t0 + t1 * t1;
#pragma unroll
  for (int off = 1; off < 32; off <<= 1) {
    s += __shfl_xor(s, off, 32);
    q += __shfl_xor(q, off, 32);
  }
  if ((tid & 31) == 0) { sbuf[tid >> 5] = s; qbuf[tid >> 5] = q; }
  __syncthreads();
  float S = 0.f, Q = 0.f;
#pragma unroll
  for (int i = 0; i < 8; ++i) { S += sbuf[i]; Q += qbuf[i]; }
  float mean = S * (1.0f / E);
  float var  = Q * (1.0f / E) - mean * mean;
  float rstd = rsqrtf(var + 1e-5f);
  float y0 = (t0 - mean) * rstd * g[tid] + be[tid];
  float y1 = (t1 - mean) * rstd * g[tid + 256] + be[tid + 256];
  yf[base + tid] = y0;
  yf[base + tid + 256] = y1;
  if (ybf) {
    ybf[base + tid] = (__bf16)y0;
    ybf[base + tid + 256] = (__bf16)y1;
  }
}

// ---------------------------------------------------------------------------
extern "C" void kernel_launch(void* const* d_in, const int* in_sizes, int n_in,
                              void* d_out, int out_size, void* d_ws, size_t ws_size,
                              hipStream_t stream) {
  using namespace cel;
  (void)in_sizes; (void)n_in; (void)out_size; (void)ws_size;

  const float* x      = (const float*)d_in[0];          // (N,B,E)
  const float* pm     = (const float*)d_in[1];          // (B,N,N)
  const unsigned char* hidden = (const unsigned char*)d_in[2]; // (B,N,N) bool
  const unsigned char* pad    = (const unsigned char*)d_in[3]; // (B,N) bool
  const float* Wp   = (const float*)d_in[4];
  const float* bp   = (const float*)d_in[5];
  const float* Wc   = (const float*)d_in[6];
  const float* bc   = (const float*)d_in[7];
  const float* Wqkv = (const float*)d_in[8];
  const float* bqkv = (const float*)d_in[9];
  const float* Wo   = (const float*)d_in[10];
  const float* bo   = (const float*)d_in[11];
  const float* W1   = (const float*)d_in[12];
  const float* b1   = (const float*)d_in[13];
  const float* W2   = (const float*)d_in[14];
  const float* b2   = (const float*)d_in[15];
  const float* ln1g = (const float*)d_in[16];
  const float* ln1b = (const float*)d_in[17];
  const float* ln2g = (const float*)d_in[18];
  const float* ln2b = (const float*)d_in[19];

  char* ws = (char*)d_ws;
  size_t off = 0;
  auto alloc = [&](size_t bytes) -> void* {
    off = (off + 255) & ~(size_t)255;
    void* p = ws + off;
    off += bytes;
    return p;
  };

  __bf16* xb    = (__bf16*)alloc((size_t)M * E * 2);
  __bf16* wqkvb = (__bf16*)alloc((size_t)3 * E * E * 2);
  __bf16* wob   = (__bf16*)alloc((size_t)E * E * 2);
  __bf16* w1b   = (__bf16*)alloc((size_t)F * E * 2);
  __bf16* w2b   = (__bf16*)alloc((size_t)E * F * 2);
  float*  pbias = (float*)alloc((size_t)Bb * H * Ns * 4);
  float*  cbias = (float*)alloc((size_t)Bb * H * Ns * 4);
  __bf16* qbuf  = (__bf16*)alloc((size_t)Bb * H * Ns * HD * 2);
  __bf16* kbuf  = (__bf16*)alloc((size_t)Bb * H * Ns * HD * 2);
  __bf16* vtbuf = (__bf16*)alloc((size_t)Bb * H * HD * Ns * 2);
  __bf16* ctxb  = (__bf16*)alloc((size_t)M * E * 2);
  float*  aout  = (float*)alloc((size_t)M * E * 4);
  float*  yf    = (float*)alloc((size_t)M * E * 4);
  __bf16* ybf   = (__bf16*)alloc((size_t)M * E * 2);
  __bf16* h1    = (__bf16*)alloc((size_t)M * F * 2);
  float*  ffout = (float*)alloc((size_t)M * E * 4);

  // fp32 -> bf16 conversions
  cvt_bf16_kernel<<<2048, 256, 0, stream>>>(x, xb, M * E);
  cvt_bf16_kernel<<<1024, 256, 0, stream>>>(Wqkv, wqkvb, 3 * E * E);
  cvt_bf16_kernel<<<512,  256, 0, stream>>>(Wo, wob, E * E);
  cvt_bf16_kernel<<<1024, 256, 0, stream>>>(W1, w1b, F * E);
  cvt_bf16_kernel<<<1024, 256, 0, stream>>>(W2, w2b, E * F);

  // per-head parent/child biases
  pcbias_kernel<<<(Bb * H * Ns) / 256, 256, 0, stream>>>(x, Wp, bp, Wc, bc, pbias, cbias);

  // QKV projection (WMMA), scatter q/k per-head, v transposed
  gemm_qkv_kernel<<<dim3(M / 32, 3), 256, 0, stream>>>(xb, wqkvb, bqkv, qbuf, kbuf, vtbuf);

  // fused attention (WMMA QK^T + bias/masks + online softmax + WMMA PV,
  // K staged to LDS with async-tensor path)
  attn_kernel<<<dim3(Ns / 128, Bb * H), 256, 0, stream>>>(qbuf, kbuf, vtbuf, pbias, cbias,
                                                          pm, hidden, pad, ctxb);

  // output projection, residual + LN1
  gemm_f32_kernel<<<dim3(M / 32, 1), 256, 0, stream>>>(ctxb, wob, bo, aout, E, E);
  ln_kernel<<<M, 256, 0, stream>>>(x, aout, ln1g, ln1b, yf, ybf);

  // FFN
  gemm_relu_kernel<<<dim3(M / 32, F / 512), 256, 0, stream>>>(ybf, w1b, b1, h1);
  gemm_f32_kernel<<<dim3(M / 32, 1), 256, 0, stream>>>(h1, w2b, b2, ffout, F, E);

  // residual + LN2 -> output
  ln_kernel<<<M, 256, 0, stream>>>(yf, ffout, ln2g, ln2b, (float*)d_out, nullptr);
}
